// MultiHeadAttentionBlock_71640054497342
// MI455X (gfx1250) — 16-bit and 8-bit matrix operand order repaired and checked by reading, compiled, not run on hardware
//
#include <hip/hip_runtime.h>
#include <hip/hip_bf16.h>
#include <math.h>


typedef __attribute__((ext_vector_type(16))) _Float16 v16h;
typedef __attribute__((ext_vector_type(8)))  _Float16 v8h;
typedef __attribute__((ext_vector_type(8)))  float    v8f;
typedef __attribute__((ext_vector_type(8)))  int      v8i;

#define NTOK   4096
#define DIN    2048
#define PDIM   16
#define HEADS  16
#define HD     128
#define SEQ    2048
#define BH     32

__global__ void k_wquant(const float* w0, const float* w1, const float* w2,
                         const float* w3, const float* w4, const float* w5,
                         signed char* wkq, float* wvq, float* alpha) {
  const int bx  = blockIdx.x;
  const int tid = threadIdx.x;
  const float* w;
  switch (bx) {
    case 0: w = w0; break; case 1: w = w1; break; case 2: w = w2; break;
    case 3: w = w3; break; case 4: w = w4; break; default: w = w5; break;
  }
  const int N = PDIM * DIN;
  float s = 0.0f;
  for (int i = tid; i < N; i += 256) s += fabsf(w[i]);
  __shared__ float red[256];
  red[tid] = s;
  __syncthreads();
  for (int st = 128; st > 0; st >>= 1) {
    if (tid < st) red[tid] += red[tid + st];
    __syncthreads();
  }
  const float al  = fmaxf(red[0] / (float)N, 1e-5f);
  const float inv = 1.0f / al;
  if (tid == 0) alpha[bx] = al;
  if (bx < 3) {
    signed char* o = wkq + bx * N;
    for (int i = tid; i < N; i += 256) {
      float v = rintf(w[i] * inv);
      v = fminf(fmaxf(v, -1.0f), 1.0f);
      o[i] = (signed char)v;
    }
  } else {
    float* o = wvq + (bx - 3) * N;
    for (int i = tid; i < N; i += 256) {
      float v = rintf(w[i] * inv);
      v = fminf(fmaxf(v, -1.0f), 1.0f);
      o[i] = v * al;
    }
  }
}

__global__ void k_xquant(const float* __restrict__ x, signed char* __restrict__ xq,
                         float* __restrict__ inv_s) {
  const int t   = blockIdx.x;
  const int tid = threadIdx.x;
  const float* row = x + (size_t)t * DIN;
  float amax = 0.0f;
  for (int i = tid; i < DIN; i += 256) amax = fmaxf(amax, fabsf(row[i]));
  __shared__ float red[256];
  red[tid] = amax;
  __syncthreads();
  for (int st = 128; st > 0; st >>= 1) {
    if (tid < st) red[tid] = fmaxf(red[tid], red[tid + st]);
    __syncthreads();
  }
  const float cl   = fmaxf(red[0], 1e-5f);
  const float s127 = 127.0f / cl;
  if (tid == 0) inv_s[t] = cl / 127.0f;
  signed char* orow = xq + (size_t)t * DIN;
  for (int i = tid; i < DIN; i += 256) {
    float v = rintf(row[i] * s127);
    v = fminf(fmaxf(v, -128.0f), 127.0f);
    orow[i] = (signed char)v;
  }
}

__global__ void k_proj1(const signed char* __restrict__ xq,
                        const signed char* __restrict__ wkq,
                        const float* __restrict__ inv_s,
                        const float* __restrict__ alpha,
                        float* __restrict__ aBuf) {
  const int lane = threadIdx.x & 31;
  const int wid  = threadIdx.x >> 5;
  const int mat  = blockIdx.y;
  const int t0   = (blockIdx.x * 4 + wid) * 16;
  const int m    = lane & 15;
  const int hi   = lane >> 4;

  const signed char* xrow = xq  + (size_t)(t0 + m) * DIN;
  const signed char* wrow = wkq + (size_t)mat * PDIM * DIN + (size_t)m * DIN;

  v8i cc;
  #pragma unroll
  for (int r = 0; r < 8; r++) cc[r] = 0;

  for (int kb = 0; kb < DIN; kb += 64) {
    v8i a, b;
    #pragma unroll
    for (int j = 0; j < 8; j++) {
      const int jj = j & 3;
      const int ao = kb + (jj & 1) * 4 + (jj >> 1) * 16 + hi * 8 + (j >> 2) * 32;
      a[j] = *(const int*)(xrow + ao);
      const int bo = kb + (jj & 1) * 4 + (jj >> 1) * 16 + hi * 8 + (j >> 2) * 32;
      b[j] = *(const int*)(wrow + bo);
    }
    cc = __builtin_amdgcn_wmma_i32_16x16x64_iu8(true, a, true, b, cc, false, false);
  }

  const float alK = alpha[mat];
  float g[8];
  #pragma unroll
  for (int r = 0; r < 8; r++) {
    const int trow = t0 + r + hi * 8;
    const float y  = (float)cc[r] * inv_s[trow] * alK;
    g[r] = 0.5f * y * (1.0f + erff(y * 0.70710678118654752f));
  }
  #pragma unroll
  for (int r = 0; r < 8; r++) {
    float ss = g[r] * g[r];
    ss += __shfl_xor(ss, 1, 32); ss += __shfl_xor(ss, 2, 32);
    ss += __shfl_xor(ss, 4, 32); ss += __shfl_xor(ss, 8, 32);
    const float nrm = fmaxf(sqrtf(ss), 1e-12f);
    const float av  = g[r] / nrm * 4.0f;
    float am = fabsf(av);
    am = fmaxf(am, __shfl_xor(am, 1, 32)); am = fmaxf(am, __shfl_xor(am, 2, 32));
    am = fmaxf(am, __shfl_xor(am, 4, 32)); am = fmaxf(am, __shfl_xor(am, 8, 32));
    const float s2 = 127.0f / fmaxf(am, 1e-5f);
    float qv = fminf(fmaxf(rintf(av * s2), -128.0f), 127.0f) / s2;
    aBuf[((size_t)mat * NTOK + (size_t)(t0 + r + hi * 8)) * PDIM + m] = qv;
  }
}

__global__ void k_proj2(const float* __restrict__ aBuf, const float* __restrict__ wvq,
                        _Float16* __restrict__ qb, _Float16* __restrict__ kb,
                        _Float16* __restrict__ vTb) {
  const int t   = blockIdx.x;
  const int mat = blockIdx.y;
  const int tid = threadIdx.x;
  __shared__ float av[PDIM];
  if (tid < PDIM) av[tid] = aBuf[((size_t)mat * NTOK + t) * PDIM + tid];
  __syncthreads();
  float areg[PDIM];
  #pragma unroll
  for (int p = 0; p < PDIM; p++) areg[p] = av[p];

  const int b = t >> 11, s = t & 2047;
  const float* wb = wvq + (size_t)mat * DIN * PDIM;
  const int o0 = tid * 8;
  #pragma unroll
  for (int i = 0; i < 8; i++) {
    const int o = o0 + i;
    const float* wr = wb + (size_t)o * PDIM;
    float y = 0.0f;
    #pragma unroll
    for (int p = 0; p < PDIM; p++) y += areg[p] * wr[p];
    const _Float16 h = (_Float16)y;
    const int hh = o >> 7, d = o & 127;
    if (mat == 0)
      qb[(((size_t)(b * HEADS + hh)) * SEQ + s) * HD + d] = h;
    else if (mat == 1)
      kb[(((size_t)(b * HEADS + hh)) * SEQ + s) * HD + d] = h;
    else
      vTb[(((size_t)(b * HEADS + hh)) * HD + d) * SEQ + s] = h;
  }
}

__device__ __forceinline__ v16h mk16(v8h lo, v8h up) {
  v16h r;
  #pragma unroll
  for (int i = 0; i < 8; i++) { r[i] = lo[i]; r[i + 8] = up[i]; }
  return r;
}

__global__ void k_attn(const _Float16* __restrict__ qb, const _Float16* __restrict__ kb,
                       const _Float16* __restrict__ vTb, float* __restrict__ out) {
  const int lane = threadIdx.x & 31;
  const int wid  = threadIdx.x >> 5;
  const int bh   = blockIdx.y;
  const int s0   = (blockIdx.x * 8 + wid) * 16;
  const int m    = lane & 15;
  const int hi   = lane >> 4;

  const _Float16* Q  = qb  + (size_t)bh * SEQ * HD;
  const _Float16* K  = kb  + (size_t)bh * SEQ * HD;
  const _Float16* VT = vTb + (size_t)bh * HD * SEQ;

  __shared__ _Float16 plds[8][16 * 32];
  _Float16* myP = plds[wid];

  v16h qf[4];
  #pragma unroll
  for (int c = 0; c < 4; c++) {
    const _Float16* qr = Q + (size_t)(s0 + m) * HD + c * 32 + hi * 8;
    qf[c] = mk16(*(const v8h*)qr, *(const v8h*)(qr + 16));
  }

  v8f acc[8];
  #pragma unroll
  for (int v = 0; v < 8; v++)
    #pragma unroll
    for (int r = 0; r < 8; r++) acc[v][r] = 0.0f;
  float mrow[8], lrow[8];
  #pragma unroll
  for (int r = 0; r < 8; r++) { mrow[r] = -1e30f; lrow[r] = 0.0f; }

  const float sscale = 0.08838834764831845f;

  for (int k0 = 0; k0 < SEQ; k0 += 32) {
    if (k0 + 32 < SEQ)
      __builtin_prefetch(K + (size_t)(k0 + 32 + m) * HD, 0, 1);

    v8f sc[2];
    #pragma unroll
    for (int g = 0; g < 2; g++) {
      #pragma unroll
      for (int r = 0; r < 8; r++) sc[g][r] = 0.0f;
      #pragma unroll
      for (int c = 0; c < 4; c++) {
        const _Float16* kr = K + (size_t)(k0 + g * 16 + m) * HD + c * 32 + hi * 8;
        v16h bf = mk16(*(const v8h*)kr, *(const v8h*)(kr + 16));
        sc[g] = __builtin_amdgcn_wmma_f32_16x16x32_f16(
            false, qf[c], false, bf, (short)0, sc[g], false, false);
      }
    }

    #pragma unroll
    for (int r = 0; r < 8; r++) {
      const float a0 = sc[0][r] * sscale;
      const float a1 = sc[1][r] * sscale;
      float mx = fmaxf(a0, a1);
      mx = fmaxf(mx, __shfl_xor(mx, 1, 32)); mx = fmaxf(mx, __shfl_xor(mx, 2, 32));
      mx = fmaxf(mx, __shfl_xor(mx, 4, 32)); mx = fmaxf(mx, __shfl_xor(mx, 8, 32));
      const float mnew = fmaxf(mrow[r], mx);
      const float corr = __expf(mrow[r] - mnew);
      mrow[r] = mnew;
      const float p0 = __expf(a0 - mnew);
      const float p1 = __expf(a1 - mnew);
      float ps = p0 + p1;
      ps += __shfl_xor(ps, 1, 32); ps += __shfl_xor(ps, 2, 32);
      ps += __shfl_xor(ps, 4, 32); ps += __shfl_xor(ps, 8, 32);
      lrow[r] = lrow[r] * corr + ps;
      #pragma unroll
      for (int v = 0; v < 8; v++) acc[v][r] *= corr;
      myP[(r + hi * 8) * 32 + m]      = (_Float16)p0;
      myP[(r + hi * 8) * 32 + 16 + m] = (_Float16)p1;
    }
    asm volatile("s_wait_dscnt 0" ::: "memory");

    const _Float16* pr = myP + m * 32 + hi * 8;
    v16h pf = mk16(*(const v8h*)pr, *(const v8h*)(myP + m * 32 + 16 + hi * 8));

    #pragma unroll
    for (int v = 0; v < 8; v++) {
      const _Float16* vr = VT + (size_t)(v * 16 + m) * SEQ + k0 + hi * 8;
      v16h bv = mk16(*(const v8h*)vr, *(const v8h*)(vr + 16));
      acc[v] = __builtin_amdgcn_wmma_f32_16x16x32_f16(
          false, pf, false, bv, (short)0, acc[v], false, false);
    }
    asm volatile("s_wait_dscnt 0" ::: "memory");
  }

  const int b = bh >> 4, h = bh & 15;
  #pragma unroll
  for (int v = 0; v < 8; v++) {
    #pragma unroll
    for (int r = 0; r < 8; r++) {
      const size_t row = (size_t)b * SEQ + s0 + r + hi * 8;
      out[row * DIN + h * HD + v * 16 + m] = acc[v][r] / lrow[r];
    }
  }
}

extern "C" void kernel_launch(void* const* d_in, const int* in_sizes, int n_in,
                              void* d_out, int out_size, void* d_ws, size_t ws_size,
                              hipStream_t stream) {
  (void)in_sizes; (void)n_in; (void)out_size; (void)ws_size;
  const float* x = (const float*)d_in[0];

  char* ws = (char*)d_ws;
  const size_t off_xq   = 0;
  const size_t off_wkq  = off_xq  + (size_t)NTOK * DIN;
  const size_t off_wvq  = off_wkq + 3ull * PDIM * DIN;
  const size_t off_invs = off_wvq + 3ull * DIN * PDIM * 4;
  const size_t off_alph = off_invs + (size_t)NTOK * 4;
  const size_t off_a    = off_alph + 256;
  const size_t off_q    = off_a    + 3ull * NTOK * PDIM * 4;
  const size_t off_k    = off_q    + 2ull * BH * SEQ * HD;
  const size_t off_vT   = off_k    + 2ull * BH * SEQ * HD;

  signed char* xq   = (signed char*)(ws + off_xq);
  signed char* wkq  = (signed char*)(ws + off_wkq);
  float*       wvq  = (float*)(ws + off_wvq);
  float*       invs = (float*)(ws + off_invs);
  float*       alph = (float*)(ws + off_alph);
  float*       aBuf = (float*)(ws + off_a);
  _Float16*    qb   = (_Float16*)(ws + off_q);
  _Float16*    kbuf = (_Float16*)(ws + off_k);
  _Float16*    vTb  = (_Float16*)(ws + off_vT);

  k_wquant<<<6, 256, 0, stream>>>((const float*)d_in[1], (const float*)d_in[3],
                                  (const float*)d_in[5], (const float*)d_in[2],
                                  (const float*)d_in[4], (const float*)d_in[6],
                                  wkq, wvq, alph);
  k_xquant<<<NTOK, 256, 0, stream>>>(x, xq, invs);
  k_proj1<<<dim3(64, 3), 128, 0, stream>>>(xq, wkq, invs, alph, aBuf);
  k_proj2<<<dim3(NTOK, 3), 256, 0, stream>>>(aBuf, wvq, qb, kbuf, vTb);
  k_attn<<<dim3(16, 32), 256, 0, stream>>>(qb, kbuf, vTb, (float*)d_out);
}
